// ClassicalQLSTM_65481071397224
// MI455X (gfx1250) — matrix operand order repaired and checked by reading, not compiled, not run on hardware
//
#include <hip/hip_runtime.h>
#include <hip/hip_bf16.h>
#include <math.h>

// ---------------------------------------------------------------------------
// LSTM on MI455X (gfx1250): bf16 WMMA for the fused 4-gate GEMM, fp32 cell.
// Async-to-LDS (ASYNCcnt) DMA for weight/h staging, double-buffered LDS.
//   inputs: [512, 128, 512] fp32, W_* : [512, 1024] fp32, b_* : [512]
//   out:    outputs [512,128,512] ++ hx [128,512] ++ cx [128,512]  (fp32)
// ---------------------------------------------------------------------------

typedef __attribute__((ext_vector_type(16))) __bf16 v16bf;
typedef __attribute__((ext_vector_type(8)))  float  v8f;

#define SEQ   512
#define BATCH 128
#define HID   512
#define KDIM  1024      // INPUT_DIM + HIDDEN_DIM
#define NDIM  2048      // 4 * HIDDEN_DIM
#define KC    64        // K-chunk
#define LPAD  72        // LDS row stride (elements), 144B = 16B-aligned

__device__ __forceinline__ __bf16 to_bf16(float f) {
  unsigned int u = __builtin_bit_cast(unsigned int, f);
  unsigned int r = u + 0x7FFFu + ((u >> 16) & 1u);   // round-to-nearest-even
  unsigned short s = (unsigned short)(r >> 16);
  return __builtin_bit_cast(__bf16, s);
}

__device__ __forceinline__ float sigf(float x) {
  return 1.0f / (1.0f + expf(-x));
}

__device__ __forceinline__ void cvt8(const float4 a, const float4 b, __bf16* d) {
  d[0] = to_bf16(a.x); d[1] = to_bf16(a.y); d[2] = to_bf16(a.z); d[3] = to_bf16(a.w);
  d[4] = to_bf16(b.x); d[5] = to_bf16(b.y); d[6] = to_bf16(b.z); d[7] = to_bf16(b.w);
}

// async DMA: 16 bytes global -> LDS, tracked by ASYNCcnt
__device__ __forceinline__ void async_b128(unsigned int lds_off, const void* gptr) {
  asm volatile("global_load_async_to_lds_b128 %0, %1, off"
               :: "v"(lds_off), "v"(gptr) : "memory");
}
__device__ __forceinline__ void wait_async0() {
  asm volatile("s_wait_asynccnt 0x0" ::: "memory");
}

// ---------------------------------------------------------------------------
// Pack W_all = concat(Wf,Wi,Wg,Wo) as bf16, row-major [n=0..2047][k=0..1023].
// ---------------------------------------------------------------------------
__global__ __launch_bounds__(256) void pack_weights(
    const float* __restrict__ Wf, const float* __restrict__ Wi,
    const float* __restrict__ Wg, const float* __restrict__ Wo,
    __bf16* __restrict__ Wc)
{
  int idx = blockIdx.x * 256 + threadIdx.x;           // < 2048*1024
  int n = idx >> 10, k = idx & 1023;
  int gate = n >> 9, row = n & 511;
  const float* src = (gate == 0) ? Wf : (gate == 1) ? Wi : (gate == 2) ? Wg : Wo;
  Wc[idx] = to_bf16(src[(size_t)row * KDIM + k]);
}

__global__ __launch_bounds__(256) void init_state(
    __bf16* __restrict__ hbf, float* __restrict__ c)
{
  int idx = blockIdx.x * 256 + threadIdx.x;           // < 128*512
  hbf[idx] = __builtin_bit_cast(__bf16, (unsigned short)0);
  c[idx] = 0.0f;
}

// ---------------------------------------------------------------------------
// Per-step gate GEMM: gates[b, n] = sum_k combined[b,k] * Wc[n,k]
//   combined[b, k<512]  = x[t, b, k]   (fp32 -> bf16 VALU path)
//   combined[b, k>=512] = h[b, k-512]  (bf16, async-to-LDS DMA)
// Block 256 threads (8 waves). Tile: 64(M) x 128(N), K-chunks of 64,
// double-buffered LDS. Wave w: M-subtile (w&3), N-subtiles (w>>2)*4..+3.
// ---------------------------------------------------------------------------
__global__ __launch_bounds__(256) void lstm_gate_gemm(
    const float*  __restrict__ xall,   // [SEQ][BATCH][512]
    const __bf16* __restrict__ Wc,     // [2048][1024]
    const __bf16* __restrict__ hbf,    // [BATCH][512]
    float*        __restrict__ gates,  // [BATCH][2048]
    int t)
{
  __shared__ __bf16 As[2][64][LPAD];
  __shared__ __bf16 Bs[2][128][LPAD];

  const int tid   = threadIdx.x;
  const int lane  = tid & 31;
  const int wave  = tid >> 5;
  const int mt    = wave & 3;
  const int ntB   = (wave >> 2) * 4;
  const int m     = lane & 15;
  const int khalf = lane >> 4;

  const int mBlock = blockIdx.y * 64;
  const int nBlock = blockIdx.x * 128;

  const v8f vzero = {0.f, 0.f, 0.f, 0.f, 0.f, 0.f, 0.f, 0.f};
  v8f acc[4];
#pragma unroll
  for (int j = 0; j < 4; ++j) acc[j] = vzero;

  // ---- staging: issues loads for K-chunk kcs into buffer `buf` ----
  auto stage = [&](int kcs, int buf) {
    const int k0 = kcs * KC;
    // A tile: 64 x 64 bf16, 256 threads x 16 elems (32B)
    {
      const int r   = tid >> 2;
      const int kk0 = (tid & 3) * 16;
      const int gb  = mBlock + r;
      const int gk  = k0 + kk0;
      if (k0 < 512) {            // x part: fp32 -> bf16 convert
        const float4* xp = reinterpret_cast<const float4*>(
            xall + ((size_t)t * BATCH + gb) * 512 + gk);
        float4 x0 = xp[0], x1 = xp[1], x2 = xp[2], x3 = xp[3];
        union { float4 f[2]; __bf16 h[16]; } pk;
        cvt8(x0, x1, pk.h);
        cvt8(x2, x3, pk.h + 8);
        *reinterpret_cast<float4*>(&As[buf][r][kk0])     = pk.f[0];
        *reinterpret_cast<float4*>(&As[buf][r][kk0 + 8]) = pk.f[1];
      } else {                   // h part: raw bf16, async DMA to LDS
        const __bf16* hp = hbf + (size_t)gb * 512 + (gk - 512);
        unsigned int l = (unsigned int)(uintptr_t)&As[buf][r][kk0];
        async_b128(l,      hp);
        async_b128(l + 16, hp + 8);
      }
    }
    // B tile: 128 x 64 bf16, 256 threads x 32 elems (64B), all async DMA
    {
      const int n   = tid >> 1;
      const int kk0 = (tid & 1) * 32;
      const __bf16* wp = Wc + (size_t)(nBlock + n) * KDIM + k0 + kk0;
      unsigned int l = (unsigned int)(uintptr_t)&Bs[buf][n][kk0];
      async_b128(l,      wp);
      async_b128(l + 16, wp + 8);
      async_b128(l + 32, wp + 16);
      async_b128(l + 48, wp + 24);
    }
  };

  auto sync_stage = [&]() {
    wait_async0();        // this wave's async-to-LDS DMAs complete
    __syncthreads();      // all waves' staging visible; prior reads done
  };

  // ---- compute: 8 WMMAs from buffer `buf` (K = 2 x 32) ----
  auto compute = [&](int buf) {
    const int arow = mt * 16 + m;
#pragma unroll
    for (int kh = 0; kh < 2; ++kh) {
      const int kb = kh * 32;
      union { v16bf v; float4 f[2]; } afr;
      afr.f[0] = *reinterpret_cast<const float4*>(&As[buf][arow][kb + khalf * 8]);
      afr.f[1] = *reinterpret_cast<const float4*>(&As[buf][arow][kb + 16 + khalf * 8]);
      union { v16bf v; float4 f[2]; } bfr[4];
#pragma unroll
      for (int j = 0; j < 4; ++j) {
        const int brow = (ntB + j) * 16 + m;   // column index n
        bfr[j].f[0] = *reinterpret_cast<const float4*>(&Bs[buf][brow][kb + khalf * 8]);
        bfr[j].f[1] = *reinterpret_cast<const float4*>(&Bs[buf][brow][kb + 16 + khalf * 8]);
      }
#pragma unroll
      for (int j = 0; j < 4; ++j) {
        acc[j] = __builtin_amdgcn_wmma_f32_16x16x32_bf16(
            false, afr.v, false, bfr[j].v, (short)0, acc[j], false, false);
      }
    }
  };

  // ---- software-pipelined main loop over 16 K-chunks ----
  stage(0, 0);
  sync_stage();
#pragma unroll 1
  for (int kc = 0; kc < KDIM / KC; ++kc) {
    const int cur = kc & 1;
    if (kc + 1 < KDIM / KC) stage(kc + 1, cur ^ 1);  // DMA overlaps WMMA burst
    compute(cur);
    if (kc + 1 < KDIM / KC) sync_stage();
  }

  // ---- write C tiles (fp32 gates, no bias yet) ----
#pragma unroll
  for (int j = 0; j < 4; ++j) {
    const int gc = nBlock + (ntB + j) * 16 + m;
#pragma unroll
    for (int v = 0; v < 8; ++v) {
      const int gr = mBlock + mt * 16 + v + 8 * khalf;
      gates[(size_t)gr * NDIM + gc] = acc[j][v];
    }
  }
}

// ---------------------------------------------------------------------------
// Per-step cell update: bias add + sigmoid/tanh + c/h update.
// ---------------------------------------------------------------------------
__global__ __launch_bounds__(256) void lstm_cell(
    const float* __restrict__ gates,                 // [BATCH][2048]
    const float* __restrict__ b_f, const float* __restrict__ b_i,
    const float* __restrict__ b_g, const float* __restrict__ b_o,
    float* __restrict__ c, __bf16* __restrict__ hbf,
    float* __restrict__ out_h)                       // [BATCH][512] slice
{
  int idx = blockIdx.x * 256 + threadIdx.x;          // < 128*512
  int b = idx >> 9, n = idx & 511;
  const float* gr = gates + (size_t)b * NDIM;
  float f = sigf(gr[n]           + b_f[n]);
  float i = sigf(gr[512  + n]    + b_i[n]);
  float g = tanhf(gr[1024 + n]   + b_g[n]);
  float o = sigf(gr[1536 + n]    + b_o[n]);
  float cn = f * c[idx] + i * g;
  float h  = o * tanhf(cn);
  c[idx]   = cn;
  hbf[idx] = to_bf16(h);
  out_h[idx] = h;
}

// ---------------------------------------------------------------------------
// Append hx (= last h) and cx to the output buffer.
// ---------------------------------------------------------------------------
__global__ __launch_bounds__(256) void finalize_out(
    float* __restrict__ out, const float* __restrict__ c)
{
  int idx = blockIdx.x * 256 + threadIdx.x;          // < 128*512
  const size_t OUT_SEQ = (size_t)SEQ * BATCH * HID;  // 33554432
  out[OUT_SEQ + idx]               = out[(size_t)(SEQ - 1) * BATCH * HID + idx];
  out[OUT_SEQ + BATCH * HID + idx] = c[idx];
}

extern "C" void kernel_launch(void* const* d_in, const int* in_sizes, int n_in,
                              void* d_out, int out_size, void* d_ws, size_t ws_size,
                              hipStream_t stream) {
  const float* xall = (const float*)d_in[0];
  const float* W_f  = (const float*)d_in[1];
  const float* b_f  = (const float*)d_in[2];
  const float* W_i  = (const float*)d_in[3];
  const float* b_i  = (const float*)d_in[4];
  const float* W_g  = (const float*)d_in[5];
  const float* b_g  = (const float*)d_in[6];
  const float* W_o  = (const float*)d_in[7];
  const float* b_o  = (const float*)d_in[8];
  float* out = (float*)d_out;

  // workspace layout
  char* ws = (char*)d_ws;
  __bf16* Wc    = (__bf16*)ws;                              // 4 MB
  float*  gates = (float*)(ws + (4u << 20));                // 1 MB
  __bf16* hbf   = (__bf16*)(ws + (5u << 20));               // 128 KB
  float*  c     = (float*)(ws + (5u << 20) + (128u << 10)); // 256 KB

  pack_weights<<<(NDIM * KDIM) / 256, 256, 0, stream>>>(W_f, W_i, W_g, W_o, Wc);
  init_state<<<(BATCH * HID) / 256, 256, 0, stream>>>(hbf, c);

  dim3 gemm_grid(NDIM / 128, BATCH / 64);   // 16 x 2 workgroups
  for (int t = 0; t < SEQ; ++t) {
    lstm_gate_gemm<<<gemm_grid, 256, 0, stream>>>(xall, Wc, hbf, gates, t);
    lstm_cell<<<(BATCH * HID) / 256, 256, 0, stream>>>(
        gates, b_f, b_i, b_g, b_o, c, hbf, out + (size_t)t * BATCH * HID);
  }
  finalize_out<<<(BATCH * HID) / 256, 256, 0, stream>>>(out, c);
}
